// SECOND_NECK_61083024884233
// MI455X (gfx1250) — matrix operand order repaired and checked by reading, not compiled, not run on hardware
//
#include <hip/hip_runtime.h>
#include <math.h>

typedef __attribute__((ext_vector_type(16))) _Float16 v16h;
typedef __attribute__((ext_vector_type(8)))  _Float16 v8h;
typedef __attribute__((ext_vector_type(8)))  float    v8f;

#define OCC 0.05f
#define BN_EPS 1e-3f
#define NB_STATS 1024

// ---------------------------------------------------------------------------
// mask init + first activation: act0 = x * mask
// ---------------------------------------------------------------------------
__global__ void init_mask_act(const float* __restrict__ x, const float* __restrict__ mraw,
                              float* __restrict__ act, float* __restrict__ mask, int nvox) {
    int v = blockIdx.x * blockDim.x + threadIdx.x;
    if (v >= nvox) return;
    float m = (mraw[v] < OCC) ? 1.f : 0.f;
    mask[v] = m;
    #pragma unroll
    for (int c = 0; c < 4; ++c) act[v * 4 + c] = x[v * 4 + c] * m;
}

// ---------------------------------------------------------------------------
// pack [wb ; ws] -> f16, layout [tap][cout][Kpad], K zero-padded to mult of 32
// ---------------------------------------------------------------------------
__global__ void pack_weights(const float* __restrict__ wb, const float* __restrict__ ws,
                             _Float16* __restrict__ wp, int cin, int cout, int Kpad) {
    int i = blockIdx.x * blockDim.x + threadIdx.x;
    int total = 27 * cout * Kpad;
    if (i >= total) return;
    int kk  = i % Kpad;
    int t2  = i / Kpad;
    int o   = t2 % cout;
    int tap = t2 / cout;
    int cin9 = cin * 9;
    float w = 0.f;
    if (kk < cin)        w = wb[((size_t)tap * cin + kk) * cout + o];
    else if (kk < cin9)  w = ws[((size_t)tap * (cin * 8) + (kk - cin)) * cout + o];
    wp[i] = (_Float16)w;
}

// ---------------------------------------------------------------------------
// zero the padded tail channels of the feature rows
// ---------------------------------------------------------------------------
__global__ void zero_pad_feat(_Float16* __restrict__ feat, int nvox, int Kpad, int cin9) {
    int padn = Kpad - cin9;
    long long i = (long long)blockIdx.x * blockDim.x + threadIdx.x;
    if (i >= (long long)nvox * padn) return;
    long long v = i / padn;
    int p = (int)(i % padn);
    feat[v * (size_t)Kpad + cin9 + p] = (_Float16)0.f;
}

// ---------------------------------------------------------------------------
// feature expansion: [silu(x)*m , bspline(x)*m] -> f16 rows of length Kpad
// Cox-de Boor, GRID_SIZE=5, order 3, knots g[i] = (i-3)*0.4 - 1, i=0..11
// ---------------------------------------------------------------------------
__global__ void expand(const float* __restrict__ act, const float* __restrict__ mask,
                       _Float16* __restrict__ feat, int nvox, int cin, int Kpad) {
    long long i = (long long)blockIdx.x * blockDim.x + threadIdx.x;
    if (i >= (long long)nvox * cin) return;
    int c = (int)(i % cin);
    long long v = i / cin;
    float x  = act[v * cin + c];
    float mk = mask[v];
    _Float16* row = feat + v * (size_t)Kpad;
    float sig = 1.f / (1.f + expf(-x));
    row[c] = (_Float16)(x * sig * mk);
    float b[11];
    #pragma unroll
    for (int j = 0; j < 11; ++j) {
        float g0 = (j - 3) * 0.4f - 1.f;
        float g1 = (j - 2) * 0.4f - 1.f;
        b[j] = (x >= g0 && x < g1) ? 1.f : 0.f;
    }
    #pragma unroll
    for (int k = 1; k <= 3; ++k) {
        for (int j = 0; j <= 10 - k; ++j) {
            float gj   = (j - 3) * 0.4f - 1.f;
            float gj1  = (j - 2) * 0.4f - 1.f;
            float gjk  = (j + k - 3) * 0.4f - 1.f;
            float gjk1 = (j + k - 2) * 0.4f - 1.f;
            b[j] = (x - gj) / (gjk - gj) * b[j] + (gjk1 - x) / (gjk1 - gj1) * b[j + 1];
        }
    }
    #pragma unroll
    for (int j = 0; j < 8; ++j) row[cin + c * 8 + j] = (_Float16)(b[j] * mk);
}

// ---------------------------------------------------------------------------
// WMMA conv helpers. A 16x32 f16: lane grp0 K 0..7,16..23 ; grp1 K 8..15,24..31.
// B 32x16 f16: lane grp0 K 0..7,16..23 ; grp1 K 8..15,24..31 (same as A).
// ---------------------------------------------------------------------------
union Frag { v16h f16; v8h h8[2]; };

__device__ __forceinline__ void load_a(Frag& a, bool vld, const _Float16* arow,
                                       int kc, int hi, const v16h& zf) {
    if (vld) {
        const _Float16* ap = arow + kc + (hi << 3);
        a.h8[0] = *(const v8h*)(ap);
        a.h8[1] = *(const v8h*)(ap + 16);
    } else {
        a.f16 = zf;
    }
}
__device__ __forceinline__ void load_b(Frag& b, const _Float16* brow, int kc, int hi) {
    const _Float16* bp = brow + kc + (hi << 3);
    b.h8[0] = *(const v8h*)(bp);
    b.h8[1] = *(const v8h*)(bp + 16);
}

// cout >= 32: one wave computes 16 voxels x 32 couts (A reused for 2 WMMAs)
__global__ void conv_wmma_n2(const _Float16* __restrict__ feat, const _Float16* __restrict__ wp,
                             float* __restrict__ y,
                             int inD, int inH, int inW, int outD, int outH, int outW,
                             int stride, int pad, int Kpad, int cout) {
    int lane = threadIdx.x & 31;
    int wave = (blockIdx.x * blockDim.x + threadIdx.x) >> 5;
    int outVox = outD * outH * outW;
    int Mtiles = (outVox + 15) >> 4;
    int Ntiles2 = cout >> 5;
    if (wave >= Mtiles * Ntiles2) return;
    int tM = wave / Ntiles2, tN = wave % Ntiles2;
    int m  = lane & 15;
    int hi = lane >> 4;
    int ovox = tM * 16 + m;
    bool mv = ovox < outVox;
    int t = mv ? ovox : 0;
    int ox = t % outW; t /= outW;
    int oy = t % outH;
    int oz = t / outH;
    size_t tapstride = (size_t)cout * Kpad;
    const _Float16* b0base = wp + ((size_t)tN * 32 + m) * Kpad;
    const _Float16* b1base = b0base + (size_t)16 * Kpad;
    v16h zf = {};
    v8f acc0 = {0.f, 0.f, 0.f, 0.f, 0.f, 0.f, 0.f, 0.f};
    v8f acc1 = acc0;
    for (int tap = 0; tap < 27; ++tap) {
        int dz = tap / 9, dy = (tap / 3) % 3, dx = tap % 3;
        int iz = oz * stride - pad + dz;
        int iy = oy * stride - pad + dy;
        int ix = ox * stride - pad + dx;
        bool vld = mv && (unsigned)iz < (unsigned)inD &&
                         (unsigned)iy < (unsigned)inH &&
                         (unsigned)ix < (unsigned)inW;
        const _Float16* arow  = feat + (size_t)((iz * inH + iy) * inW + ix) * Kpad;
        const _Float16* brow0 = b0base + (size_t)tap * tapstride;
        const _Float16* brow1 = b1base + (size_t)tap * tapstride;
        if (tap < 26) {  // prefetch next tap's weight rows (global_prefetch_b8)
            __builtin_prefetch(brow0 + tapstride, 0, 1);
            __builtin_prefetch(brow1 + tapstride, 0, 1);
        }
        for (int kc = 0; kc < Kpad; kc += 32) {
            Frag a, b0, b1;
            load_a(a, vld, arow, kc, hi, zf);
            load_b(b0, brow0, kc, hi);
            load_b(b1, brow1, kc, hi);
            acc0 = __builtin_amdgcn_wmma_f32_16x16x32_f16(false, a.f16, false, b0.f16,
                                                          (short)0, acc0, false, false);
            acc1 = __builtin_amdgcn_wmma_f32_16x16x32_f16(false, a.f16, false, b1.f16,
                                                          (short)0, acc1, false, false);
        }
    }
    #pragma unroll
    for (int r = 0; r < 8; ++r) {  // D: row = r + 8*hi, col = lane&15
        int sv = tM * 16 + r + (hi << 3);
        if (sv < outVox) {
            float* yp = y + (size_t)sv * cout + tN * 32 + m;
            yp[0]  = acc0[r];
            yp[16] = acc1[r];
        }
    }
}

// cout == 16: one wave computes 32 voxels x 16 couts (B reused for 2 WMMAs)
__global__ void conv_wmma_m2(const _Float16* __restrict__ feat, const _Float16* __restrict__ wp,
                             float* __restrict__ y,
                             int inD, int inH, int inW, int outD, int outH, int outW,
                             int stride, int pad, int Kpad, int cout) {
    int lane = threadIdx.x & 31;
    int wave = (blockIdx.x * blockDim.x + threadIdx.x) >> 5;
    int outVox = outD * outH * outW;
    int Mtiles2 = (outVox + 31) >> 5;
    if (wave >= Mtiles2) return;
    int tM = wave;
    int m  = lane & 15;
    int hi = lane >> 4;
    int ovox0 = tM * 32 + m;
    int ovox1 = ovox0 + 16;
    bool mv0 = ovox0 < outVox, mv1 = ovox1 < outVox;
    int t0 = mv0 ? ovox0 : 0;
    int ox0 = t0 % outW; t0 /= outW;
    int oy0 = t0 % outH; int oz0 = t0 / outH;
    int t1 = mv1 ? ovox1 : 0;
    int ox1 = t1 % outW; t1 /= outW;
    int oy1 = t1 % outH; int oz1 = t1 / outH;
    size_t tapstride = (size_t)cout * Kpad;
    const _Float16* bbase = wp + (size_t)m * Kpad;
    v16h zf = {};
    v8f acc0 = {0.f, 0.f, 0.f, 0.f, 0.f, 0.f, 0.f, 0.f};
    v8f acc1 = acc0;
    for (int tap = 0; tap < 27; ++tap) {
        int dz = tap / 9, dy = (tap / 3) % 3, dx = tap % 3;
        int iz0 = oz0 * stride - pad + dz, iy0 = oy0 * stride - pad + dy, ix0 = ox0 * stride - pad + dx;
        int iz1 = oz1 * stride - pad + dz, iy1 = oy1 * stride - pad + dy, ix1 = ox1 * stride - pad + dx;
        bool vld0 = mv0 && (unsigned)iz0 < (unsigned)inD && (unsigned)iy0 < (unsigned)inH &&
                           (unsigned)ix0 < (unsigned)inW;
        bool vld1 = mv1 && (unsigned)iz1 < (unsigned)inD && (unsigned)iy1 < (unsigned)inH &&
                           (unsigned)ix1 < (unsigned)inW;
        const _Float16* arow0 = feat + (size_t)((iz0 * inH + iy0) * inW + ix0) * Kpad;
        const _Float16* arow1 = feat + (size_t)((iz1 * inH + iy1) * inW + ix1) * Kpad;
        const _Float16* brow  = bbase + (size_t)tap * tapstride;
        if (tap < 26) {  // prefetch next tap's weight row (global_prefetch_b8)
            __builtin_prefetch(brow + tapstride, 0, 1);
        }
        for (int kc = 0; kc < Kpad; kc += 32) {
            Frag a0, a1, b;
            load_b(b, brow, kc, hi);
            load_a(a0, vld0, arow0, kc, hi, zf);
            load_a(a1, vld1, arow1, kc, hi, zf);
            acc0 = __builtin_amdgcn_wmma_f32_16x16x32_f16(false, a0.f16, false, b.f16,
                                                          (short)0, acc0, false, false);
            acc1 = __builtin_amdgcn_wmma_f32_16x16x32_f16(false, a1.f16, false, b.f16,
                                                          (short)0, acc1, false, false);
        }
    }
    #pragma unroll
    for (int r = 0; r < 8; ++r) {
        int sv0 = tM * 32 + r + (hi << 3);
        int sv1 = sv0 + 16;
        if (sv0 < outVox) y[(size_t)sv0 * cout + m] = acc0[r];
        if (sv1 < outVox) y[(size_t)sv1 * cout + m] = acc1[r];
    }
}

// ---------------------------------------------------------------------------
// mask dilation for non-submanifold (strided) blocks: omask = conv(mask,1) > 0
// ---------------------------------------------------------------------------
__global__ void dilate_mask(const float* __restrict__ mi, float* __restrict__ mo,
                            int inD, int inH, int inW, int outD, int outH, int outW,
                            int stride, int pad) {
    int i = blockIdx.x * blockDim.x + threadIdx.x;
    int outVox = outD * outH * outW;
    if (i >= outVox) return;
    int ox = i % outW, t = i / outW;
    int oy = t % outH, oz = t / outH;
    float s = 0.f;
    for (int dz = 0; dz < 3; ++dz) {
        int iz = oz * stride - pad + dz;
        if ((unsigned)iz >= (unsigned)inD) continue;
        for (int dy = 0; dy < 3; ++dy) {
            int iy = oy * stride - pad + dy;
            if ((unsigned)iy >= (unsigned)inH) continue;
            for (int dx = 0; dx < 3; ++dx) {
                int ix = ox * stride - pad + dx;
                if ((unsigned)ix >= (unsigned)inW) continue;
                s += mi[(iz * inH + iy) * inW + ix];
            }
        }
    }
    mo[i] = (s > 0.f) ? 1.f : 0.f;
}

// ---------------------------------------------------------------------------
// masked BN stats: per-block partials (deterministic), then fixed-order reduce
// part layout: [block][2*cout+1] = {sum[c], sumsq[c], cnt}
// ---------------------------------------------------------------------------
__global__ void bn_stats(const float* __restrict__ y, const float* __restrict__ om,
                         float* __restrict__ part, int nvox, int cout) {
    __shared__ float s1[256], s2[256], s3[256];
    int tid = threadIdx.x;
    int c = tid & (cout - 1);
    int g = tid / cout;
    int vper = 256 / cout;
    float sum = 0.f, sq = 0.f, cnt = 0.f;
    for (long long v = (long long)blockIdx.x * vper + g; v < nvox;
         v += (long long)gridDim.x * vper) {
        float mk  = om[v];
        float val = y[v * cout + c];
        sum += val * mk;
        sq  += val * val * mk;
        if (c == 0) cnt += mk;
    }
    s1[tid] = sum; s2[tid] = sq; s3[tid] = cnt;
    __syncthreads();
    for (int off = 128; off >= cout; off >>= 1) {
        if (tid < off) { s1[tid] += s1[tid + off]; s2[tid] += s2[tid + off]; s3[tid] += s3[tid + off]; }
        __syncthreads();
    }
    int n = 2 * cout + 1;
    float* p = part + (size_t)blockIdx.x * n;
    if (tid < cout) { p[tid] = s1[tid]; p[cout + tid] = s2[tid]; }
    if (tid == 0)   { p[2 * cout] = s3[0]; }
}

__global__ void bn_reduce(const float* __restrict__ part, float* __restrict__ stats,
                          int cout, int nb) {
    int t = threadIdx.x;
    int n = 2 * cout + 1;
    if (t >= n) return;
    float s = 0.f;
    for (int b = 0; b < nb; ++b) s += part[(size_t)b * n + t];
    stats[t] = s;
}

// ---------------------------------------------------------------------------
// BN + ReLU + mask (in-place safe). Last block writes d_out directly.
// ---------------------------------------------------------------------------
__global__ void bn_apply(const float* __restrict__ y, const float* __restrict__ om,
                         const float* __restrict__ stats, const float* __restrict__ gamma,
                         const float* __restrict__ beta, float* __restrict__ out,
                         int nvox, int cout) {
    long long i = (long long)blockIdx.x * blockDim.x + threadIdx.x;
    if (i >= (long long)nvox * cout) return;
    int c = (int)(i % cout);
    long long v = i / cout;
    float cnt = stats[2 * cout]; cnt = cnt < 1.f ? 1.f : cnt;
    float mean = stats[c] / cnt;
    float var  = stats[cout + c] / cnt - mean * mean;
    var = var > 0.f ? var : 0.f;
    float yn = (y[i] - mean) * rsqrtf(var + BN_EPS) * gamma[c] + beta[c];
    yn = yn > 0.f ? yn : 0.f;
    out[i] = yn * om[v];
}

// ---------------------------------------------------------------------------
extern "C" void kernel_launch(void* const* d_in, const int* in_sizes, int n_in,
                              void* d_out, int out_size, void* d_ws, size_t ws_size,
                              hipStream_t stream) {
    (void)in_sizes; (void)n_in; (void)out_size; (void)ws_size;
    const float* x    = (const float*)d_in[0];
    const float* mraw = (const float*)d_in[1];

    struct LC { int cin, cout, stride, pad, man; };
    static const LC cfg[12] = {
        {4, 16, 1, 1, 1}, {16, 16, 1, 1, 1},
        {16, 32, 2, 1, 0}, {32, 32, 1, 1, 1}, {32, 32, 1, 1, 1},
        {32, 64, 2, 1, 0}, {64, 64, 1, 1, 1}, {64, 64, 1, 1, 1},
        {64, 64, 2, 1, 0}, {64, 64, 1, 1, 1}, {64, 64, 1, 1, 1},
        {64, 128, 2, 0, 0}};

    // ---- workspace carve-up (256B aligned) ----
    char* ws = (char*)d_ws;
    size_t off = 0;
    auto carve = [&](size_t bytes) -> char* {
        char* p = ws + off;
        off = (off + bytes + 255) & ~(size_t)255;
        return p;
    };
    const size_t MAXVOX = 21 * 160 * 160;  // 537600
    float*    actA  = (float*)carve(MAXVOX * 16 * sizeof(float));
    float*    actB  = (float*)carve(MAXVOX * 16 * sizeof(float));
    float*    maskA = (float*)carve(MAXVOX * sizeof(float));
    float*    maskB = (float*)carve(MAXVOX * sizeof(float));
    _Float16* feat  = (_Float16*)carve(MAXVOX * 160 * sizeof(_Float16));
    float*    part  = (float*)carve((size_t)NB_STATS * 257 * sizeof(float));
    float*    stats = (float*)carve(257 * sizeof(float));
    _Float16* wpack[12];
    for (int i = 0; i < 12; ++i) {
        int Kpad = (cfg[i].cin * 9 + 31) & ~31;
        wpack[i] = (_Float16*)carve((size_t)27 * cfg[i].cout * Kpad * sizeof(_Float16));
    }

    // ---- weight packing (all blocks) ----
    for (int i = 0; i < 12; ++i) {
        int Kpad = (cfg[i].cin * 9 + 31) & ~31;
        int total = 27 * cfg[i].cout * Kpad;
        pack_weights<<<(total + 255) / 256, 256, 0, stream>>>(
            (const float*)d_in[2 + 4 * i], (const float*)d_in[3 + 4 * i],
            wpack[i], cfg[i].cin, cfg[i].cout, Kpad);
    }

    int D = 21, H = 160, W = 160;
    int nvox0 = D * H * W;
    init_mask_act<<<(nvox0 + 255) / 256, 256, 0, stream>>>(x, mraw, actA, maskA, nvox0);

    float* actCur = actA;  float* actNext = actB;
    float* maskCur = maskA; float* maskNext = maskB;

    for (int i = 0; i < 12; ++i) {
        const LC& L = cfg[i];
        int cin9 = L.cin * 9;
        int Kpad = (cin9 + 31) & ~31;
        int inD = D, inH = H, inW = W;
        int outD, outH, outW;
        if (L.stride == 1) { outD = inD; outH = inH; outW = inW; }
        else {
            outD = (inD + 2 * L.pad - 3) / 2 + 1;
            outH = (inH + 2 * L.pad - 3) / 2 + 1;
            outW = (inW + 2 * L.pad - 3) / 2 + 1;
        }
        int inVox = inD * inH * inW, outVox = outD * outH * outW;

        if (Kpad > cin9) {
            long long tot = (long long)inVox * (Kpad - cin9);
            zero_pad_feat<<<(unsigned)((tot + 255) / 256), 256, 0, stream>>>(feat, inVox, Kpad, cin9);
        }
        {
            long long tot = (long long)inVox * L.cin;
            expand<<<(unsigned)((tot + 255) / 256), 256, 0, stream>>>(actCur, maskCur, feat,
                                                                      inVox, L.cin, Kpad);
        }
        if (L.cout >= 32) {
            int tiles = ((outVox + 15) / 16) * (L.cout / 32);
            conv_wmma_n2<<<(tiles + 3) / 4, 128, 0, stream>>>(feat, wpack[i], actNext,
                                                              inD, inH, inW, outD, outH, outW,
                                                              L.stride, L.pad, Kpad, L.cout);
        } else {
            int tiles = (outVox + 31) / 32;
            conv_wmma_m2<<<(tiles + 3) / 4, 128, 0, stream>>>(feat, wpack[i], actNext,
                                                              inD, inH, inW, outD, outH, outW,
                                                              L.stride, L.pad, Kpad, L.cout);
        }
        float* omask = maskCur;
        if (!L.man) {
            dilate_mask<<<(outVox + 255) / 256, 256, 0, stream>>>(maskCur, maskNext,
                                                                  inD, inH, inW,
                                                                  outD, outH, outW,
                                                                  L.stride, L.pad);
            omask = maskNext;
            float* tm = maskCur; maskCur = maskNext; maskNext = tm;
        }
        bn_stats<<<NB_STATS, 256, 0, stream>>>(actNext, omask, part, outVox, L.cout);
        bn_reduce<<<1, 512, 0, stream>>>(part, stats, L.cout, NB_STATS);
        float* dst = (i == 11) ? (float*)d_out : actNext;
        {
            long long tot = (long long)outVox * L.cout;
            bn_apply<<<(unsigned)((tot + 255) / 256), 256, 0, stream>>>(
                actNext, omask, stats,
                (const float*)d_in[4 + 4 * i], (const float*)d_in[5 + 4 * i],
                dst, outVox, L.cout);
        }
        float* ta = actCur; actCur = actNext; actNext = ta;
        D = outD; H = outH; W = outW;
    }
}
